// FEDformerEncoderLayer_85478439125303
// MI455X (gfx1250) — matrix operand order repaired and checked by reading, not compiled, not run on hardware
//
#include <hip/hip_runtime.h>
#include <math.h>

// MI455X / gfx1250, wave32, WMMA f32_16x16x32_f16 path.
// FEDformer encoder layer:
//   q = x@WqT+bq  (k,v projections are dead code in the reference -> skipped)
//   Xsel = truncated DFT over L at 64 selected modes (GEMM vs cos/-sin basis)
//   Y = per-(h,m) complex ExE channel mix
//   y = truncated inverse DFT (GEMM vs weighted basis)
//   u = x + y@WoT+bo ; r1 = decomp(u) ; z = r1 + gelu(r1@c1T)@c2T ; out = decomp(z)
// GEMMs: 128x128 block tile, K-step 32, 8 waves, double-buffered LDS
// (one barrier per K-step, global loads for tile k+1 issued before the WMMAs
// of tile k so HBM latency overlaps matrix math).

typedef _Float16 v16h __attribute__((ext_vector_type(16)));
typedef _Float16 h8   __attribute__((ext_vector_type(8)));
typedef float    v8f  __attribute__((ext_vector_type(8)));
typedef float    f4   __attribute__((ext_vector_type(4)));

#define B_   16
#define L_   2048
#define D_   512
#define H_   8
#define E_   64
#define M_   64
#define DFF_ 2048

__device__ __forceinline__ v16h hcombine(h8 lo, h8 hi) {
  union { v16h v; h8 h[2]; } u;
  u.h[0] = lo; u.h[1] = hi;
  return u.v;
}

__device__ __forceinline__ void cvt2h8(f4 a0, f4 a1, f4 a2, f4 a3, h8& o0, h8& o1) {
#pragma unroll
  for (int i = 0; i < 4; ++i) {
    o0[i]     = (_Float16)a0[i];
    o0[4 + i] = (_Float16)a1[i];
    o1[i]     = (_Float16)a2[i];
    o1[4 + i] = (_Float16)a3[i];
  }
}

// ---------------------------------------------------------------------------
// Generic WMMA GEMM: C[M,N] = epi(A[M,K] * B[N,K]^T + bias) (+ residual)
//  - A source: f32 or f16; row-major (element (row,k) at A[row*lda + k]) or
//    KMAJOR (element (row,k) at A[k*lda + row]) for the transposed DFT input.
//  - B source: always f32, element (n,k) at Bsrc[n*ldb + k] (k contiguous).
//  - EPI: 0 = none, 1 = exact GELU.
// All problem sizes here are multiples of the tiles -> no bounds checks.
// ---------------------------------------------------------------------------
template <typename AT, typename OT, bool KMAJOR, int EPI>
__global__ void __launch_bounds__(256) gemm_wmma(
    const AT* __restrict__ A, const float* __restrict__ Bsrc,
    const float* __restrict__ bias, const float* __restrict__ residual,
    OT* __restrict__ C, int Kdim,
    long sAb, int lda, long sBb, int ldb, long sCb, int ldc)
{
  __shared__ _Float16 la[2][128 * 40];  // pitch 40 f16 = 80B: conflict-free b128
  __shared__ _Float16 lb[2][128 * 40];

  const int t    = threadIdx.x;
  const int ln   = t & 31;
  const int wid  = t >> 5;
  const int half = ln >> 4;
  const int lr   = ln & 15;
  const int wrow = wid & 3;            // 4 wave-rows of 32
  const int wcol = wid >> 2;           // 2 wave-cols of 64
  const int row0 = blockIdx.y * 128;
  const int col0 = blockIdx.x * 128;

  const AT*    Ab = A + (long)blockIdx.z * sAb;
  const float* Bb = Bsrc + (long)blockIdx.z * sBb;
  OT*          Cb = C + (long)blockIdx.z * sCb;
  const float* Rb = residual ? (residual + (long)blockIdx.z * sCb) : nullptr;

  // staging thread mapping
  const int rA   = t >> 1;             // row-major A row / B row (0..127)
  const int khA  = (t & 1) << 4;       // 0 or 16
  const int lsub = t >> 3;             // KMAJOR: k within tile (0..31)
  const int dg   = (t & 7) << 4;       // KMAJOR: 16-row group (0..112)

  // ---- prologue: stage K-tile 0 into buffer 0 ----
  {
    if constexpr (KMAJOR) {
      const _Float16* s0 = (const _Float16*)Ab + (long)lsub * lda + row0 + dg;
      h8 v0 = *(const h8*)s0;
      h8 v1 = *(const h8*)(s0 + 8);
#pragma unroll
      for (int i = 0; i < 8; ++i) la[0][(dg + i) * 40 + lsub] = v0[i];
#pragma unroll
      for (int i = 0; i < 8; ++i) la[0][(dg + 8 + i) * 40 + lsub] = v1[i];
    } else {
      const AT* s0 = Ab + (long)(row0 + rA) * lda + khA;
      h8 o0, o1;
      if constexpr (sizeof(AT) == 4) {
        cvt2h8(*(const f4*)s0, *(const f4*)(s0 + 4),
               *(const f4*)(s0 + 8), *(const f4*)(s0 + 12), o0, o1);
      } else {
        o0 = *(const h8*)s0;
        o1 = *(const h8*)(s0 + 8);
      }
      *(h8*)&la[0][rA * 40 + khA]     = o0;
      *(h8*)&la[0][rA * 40 + khA + 8] = o1;
    }
    const float* s1 = Bb + (long)(col0 + rA) * ldb + khA;
    h8 q0, q1;
    cvt2h8(*(const f4*)s1, *(const f4*)(s1 + 4),
           *(const f4*)(s1 + 8), *(const f4*)(s1 + 12), q0, q1);
    *(h8*)&lb[0][rA * 40 + khA]     = q0;
    *(h8*)&lb[0][rA * 40 + khA + 8] = q1;
  }
  __syncthreads();

  v8f acc[2][4] = {};
  int cur = 0;

  for (int k0 = 0; k0 < Kdim; k0 += 32) {
    const bool hn = (k0 + 32) < Kdim;

    // ---- (1) issue next K-tile global loads (overlap with the WMMAs) ----
    h8 ra0{}, ra1{};
    f4 fa0{}, fa1{}, fa2{}, fa3{};
    f4 fb0{}, fb1{}, fb2{}, fb3{};
    if (hn) {
      const int kn = k0 + 32;
      if constexpr (KMAJOR) {
        const _Float16* s0 =
            (const _Float16*)Ab + (long)(kn + lsub) * lda + row0 + dg;
        ra0 = *(const h8*)s0;
        ra1 = *(const h8*)(s0 + 8);
      } else {
        const AT* s0 = Ab + (long)(row0 + rA) * lda + kn + khA;
        if constexpr (sizeof(AT) == 4) {
          fa0 = *(const f4*)s0;
          fa1 = *(const f4*)(s0 + 4);
          fa2 = *(const f4*)(s0 + 8);
          fa3 = *(const f4*)(s0 + 12);
        } else {
          ra0 = *(const h8*)s0;
          ra1 = *(const h8*)(s0 + 8);
        }
        if (kn + 32 < Kdim)
          __builtin_prefetch((const void*)(s0 + 32), 0, 3);
      }
      const float* s1 = Bb + (long)(col0 + rA) * ldb + kn + khA;
      fb0 = *(const f4*)s1;
      fb1 = *(const f4*)(s1 + 4);
      fb2 = *(const f4*)(s1 + 8);
      fb3 = *(const f4*)(s1 + 12);
      if (kn + 32 < Kdim)
        __builtin_prefetch((const void*)(s1 + 32), 0, 3);
    }

    // ---- (2) fragments from current buffer + 2x4 WMMA ----
    // A 16x32 f16 layout: lane(lr,half): row=lr; elems e0..7 -> K=half*8+e,
    // e8..15 -> K=16+half*8+e  => two contiguous 8-elem runs.
    v16h af[2], bf[4];
#pragma unroll
    for (int i = 0; i < 2; ++i) {
      const int arow = wrow * 32 + i * 16 + lr;
      af[i] = hcombine(*(const h8*)&la[cur][arow * 40 + half * 8],
                       *(const h8*)&la[cur][arow * 40 + 16 + half * 8]);
    }
    // B 32x16 f16 layout: lane: col=lr; elems e0..7 -> K=half*8+e, e8..15 -> K=16+half*8+e (same as A).
#pragma unroll
    for (int j = 0; j < 4; ++j) {
      const int bn = wcol * 64 + j * 16 + lr;
      bf[j] = hcombine(*(const h8*)&lb[cur][bn * 40 + half * 8],
                       *(const h8*)&lb[cur][bn * 40 + 16 + half * 8]);
    }
#pragma unroll
    for (int i = 0; i < 2; ++i)
#pragma unroll
      for (int j = 0; j < 4; ++j)
        acc[i][j] = __builtin_amdgcn_wmma_f32_16x16x32_f16(
            false, af[i], false, bf[j], (short)0, acc[i][j], false, false);

    // ---- (3) store next tile into the other buffer, single barrier ----
    if (hn) {
      const int ns = cur ^ 1;
      if constexpr (KMAJOR) {
#pragma unroll
        for (int i = 0; i < 8; ++i) la[ns][(dg + i) * 40 + lsub] = ra0[i];
#pragma unroll
        for (int i = 0; i < 8; ++i) la[ns][(dg + 8 + i) * 40 + lsub] = ra1[i];
      } else {
        h8 o0, o1;
        if constexpr (sizeof(AT) == 4) {
          cvt2h8(fa0, fa1, fa2, fa3, o0, o1);
        } else {
          o0 = ra0;
          o1 = ra1;
        }
        *(h8*)&la[ns][rA * 40 + khA]     = o0;
        *(h8*)&la[ns][rA * 40 + khA + 8] = o1;
      }
      h8 q0, q1;
      cvt2h8(fb0, fb1, fb2, fb3, q0, q1);
      *(h8*)&lb[ns][rA * 40 + khA]     = q0;
      *(h8*)&lb[ns][rA * 40 + khA + 8] = q1;
      __syncthreads();
    }
    cur ^= 1;
  }

  // ---- epilogue: C/D layout: lane: col=lr; vgpr v -> row = v + half*8 ----
#pragma unroll
  for (int i = 0; i < 2; ++i) {
#pragma unroll
    for (int j = 0; j < 4; ++j) {
      const int cg = col0 + wcol * 64 + j * 16 + lr;
      const float bv = bias ? bias[cg] : 0.0f;
#pragma unroll
      for (int v = 0; v < 8; ++v) {
        const int rg = row0 + wrow * 32 + i * 16 + half * 8 + v;
        float val = acc[i][j][v] + bv;
        if constexpr (EPI == 1)
          val = 0.5f * val * (1.0f + erff(val * 0.70710678118654752f));
        if (Rb) val += Rb[(long)rg * ldc + cg];
        Cb[(long)rg * ldc + cg] = (OT)val;
      }
    }
  }
}

// ---------------------------------------------------------------------------
// DFT bases. Fb[128][L]: rows 0..63 = cos(2*pi*f_m*l/L), rows 64..127 = -sin.
// Ib[L][128]: cols 0..63 = w_m*cos/L, cols 64..127 = -w_m*sin/L  (w=1 if f==0
// else 2; Nyquist never selected since modes come from range(L/2)).
// Integer-mod phase keeps full precision for large f*l.
// ---------------------------------------------------------------------------
__global__ void __launch_bounds__(256) basis_kernel(
    const int* __restrict__ mode_index, float* __restrict__ Fb,
    float* __restrict__ Ib)
{
  const int idx = blockIdx.x * 256 + threadIdx.x;  // 2048*64 threads
  const int m = idx & 63;
  const int l = idx >> 6;
  const int f = mode_index[m];
  const int r = (f * l) & (L_ - 1);
  const float ang = (float)r * 0.003067961575771282340f;  // 2*pi/2048
  float s, c;
  __sincosf(ang, &s, &c);
  Fb[m * L_ + l]        = c;
  Fb[(64 + m) * L_ + l] = -s;
  const float inv = ((f == 0) ? 1.0f : 2.0f) * (1.0f / (float)L_);
  Ib[l * 128 + m]      = c * inv;
  Ib[l * 128 + 64 + m] = -s * inv;
}

// ---------------------------------------------------------------------------
// Per-(h,m) complex ExE mix: Y[b,h,o,m] = sum_e X[b,h,e,m] * (wr+j*wi)[h,e,o,m]
// X/Y layout: [B, D, 128] with cols 0..63 = real, 64..127 = imag.
// ---------------------------------------------------------------------------
__global__ void __launch_bounds__(256) cmix_kernel(
    const float* __restrict__ X, const float* __restrict__ wr,
    const float* __restrict__ wi, float* __restrict__ Y)
{
  const int idx = blockIdx.x * 256 + threadIdx.x;  // B*D*M = 524288
  const int m = idx & 63;
  const int o = (idx >> 6) & 63;
  const int h = (idx >> 12) & 7;
  const int b = idx >> 15;
  const float* xb  = X + (long)(b * D_ + h * E_) * 128 + m;    // e-stride 128
  const float* wrp = wr + ((long)(h * E_) * E_ + o) * M_ + m;  // e-stride 4096
  const float* wip = wi + ((long)(h * E_) * E_ + o) * M_ + m;
  float yr = 0.f, yi = 0.f;
#pragma unroll 4
  for (int e = 0; e < E_; ++e) {
    const float xr = xb[e * 128];
    const float xi = xb[e * 128 + 64];
    const float a = wrp[(long)e * 4096];
    const float c = wip[(long)e * 4096];
    yr += xr * a - xi * c;
    yi += xr * c + xi * a;
  }
  const long od = (long)(b * D_ + h * E_ + o) * 128 + m;
  Y[od]      = yr;
  Y[od + 64] = yi;
}

// ---------------------------------------------------------------------------
// MoE series decomposition: out = u - sum_k softmax_k(u*w+b) * movavg_k(u),
// kernels 13 and 25, replicate padding. LDS tile: 64 L x 64 D with 12-halo.
// ---------------------------------------------------------------------------
__global__ void __launch_bounds__(256) decomp_kernel(
    const float* __restrict__ U, const float* __restrict__ gw,
    const float* __restrict__ gb, float* __restrict__ R)
{
  __shared__ float tile[88 * 65];
  const int b  = blockIdx.z;
  const int l0 = blockIdx.y * 64;
  const int d0 = blockIdx.x * 64;
  const int t  = threadIdx.x;
  const int d  = t & 63;
  const int r  = t >> 6;  // 0..3
  const float* Ub = U + (long)b * L_ * D_;
  for (int rr = r; rr < 88; rr += 4) {
    int gl = l0 - 12 + rr;
    gl = gl < 0 ? 0 : (gl > L_ - 1 ? L_ - 1 : gl);
    tile[rr * 65 + d] = Ub[(long)gl * D_ + d0 + d];
  }
  __syncthreads();
  const float w0 = gw[0], w1 = gw[1], bb0 = gb[0], bb1 = gb[1];
  for (int rr = r; rr < 64; rr += 4) {
    float s25 = 0.f, s13 = 0.f;
#pragma unroll
    for (int j = 0; j < 25; ++j) {
      const float v = tile[(rr + j) * 65 + d];
      s25 += v;
      if (j >= 6 && j <= 18) s13 += v;
    }
    const float x = tile[(rr + 12) * 65 + d];
    const float a13 = s13 * (1.0f / 13.0f);
    const float a25 = s25 * (1.0f / 25.0f);
    const float g0 = x * w0 + bb0;
    const float g1 = x * w1 + bb1;
    const float mx = fmaxf(g0, g1);
    const float e0 = __expf(g0 - mx);
    const float e1 = __expf(g1 - mx);
    const float mean = (e0 * a13 + e1 * a25) / (e0 + e1);
    R[((long)b * L_ + l0 + rr) * D_ + d0 + d] = x - mean;
  }
}

// ---------------------------------------------------------------------------
extern "C" void kernel_launch(void* const* d_in, const int* in_sizes, int n_in,
                              void* d_out, int out_size, void* d_ws,
                              size_t ws_size, hipStream_t stream)
{
  (void)in_sizes; (void)n_in; (void)out_size; (void)ws_size;
  const float* x   = (const float*)d_in[0];
  const float* Wq  = (const float*)d_in[1];
  const float* bq  = (const float*)d_in[2];
  // d_in[3..6] = Wk,bk,Wv,bv: dead in the reference, skipped.
  const float* Wo  = (const float*)d_in[7];
  const float* bo  = (const float*)d_in[8];
  const float* fwr = (const float*)d_in[9];
  const float* fwi = (const float*)d_in[10];
  const float* c1  = (const float*)d_in[11];
  const float* c2  = (const float*)d_in[12];
  const float* d1w = (const float*)d_in[13];
  const float* d1b = (const float*)d_in[14];
  const float* d2w = (const float*)d_in[15];
  const float* d2b = (const float*)d_in[16];
  const int*  midx = (const int*)d_in[17];

  char* wsp = (char*)d_ws;
  size_t off = 0;
  auto alloc = [&](size_t bytes) {
    void* p = wsp + off;
    off += (bytes + 255) & ~(size_t)255;
    return p;
  };
  _Float16* qbuf = (_Float16*)alloc((size_t)B_ * L_ * D_ * 2);   // q, later y
  float*    Fb   = (float*)alloc((size_t)128 * L_ * 4);
  float*    Ib   = (float*)alloc((size_t)L_ * 128 * 4);
  float*    X    = (float*)alloc((size_t)B_ * D_ * 128 * 4);
  float*    Y    = (float*)alloc((size_t)B_ * D_ * 128 * 4);
  float*    u    = (float*)alloc((size_t)B_ * L_ * D_ * 4);      // u, later z
  float*    r1   = (float*)alloc((size_t)B_ * L_ * D_ * 4);
  _Float16* h1   = (_Float16*)alloc((size_t)B_ * L_ * DFF_ * 2); // ~298MB total

  // 0) DFT bases at the 64 selected modes
  basis_kernel<<<512, 256, 0, stream>>>(midx, Fb, Ib);

  // 1) q = x @ Wq^T + bq   [32768,512]x[512,512] -> f16
  gemm_wmma<float, _Float16, false, 0><<<dim3(4, 256, 1), 256, 0, stream>>>(
      x, Wq, bq, nullptr, qbuf, 512, 0, 512, 0, 512, 0, 512);

  // 2) forward truncated DFT per batch: X[b] = q[b]^T [512,2048] @ Fb^T -> [512,128]
  gemm_wmma<_Float16, float, true, 0><<<dim3(1, 4, 16), 256, 0, stream>>>(
      qbuf, Fb, nullptr, nullptr, X, L_,
      (long)L_ * D_, D_, 0, L_, (long)D_ * 128, 128);

  // 3) per-(h,m) complex channel mix
  cmix_kernel<<<2048, 256, 0, stream>>>(X, fwr, fwi, Y);

  // 4) inverse truncated DFT: y[b] = Ib [2048,128] @ Y[b]^T -> [2048,512] f16
  //    (writes into qbuf, which is dead after step 2)
  gemm_wmma<float, _Float16, false, 0><<<dim3(4, 16, 16), 256, 0, stream>>>(
      Ib, Y, nullptr, nullptr, qbuf, 128,
      0, 128, (long)D_ * 128, 128, (long)L_ * D_, 512);

  // 5) u = x + y @ Wo^T + bo
  gemm_wmma<_Float16, float, false, 0><<<dim3(4, 256, 1), 256, 0, stream>>>(
      qbuf, Wo, bo, x, u, 512, 0, 512, 0, 512, 0, 512);

  // 6) r1 = decomp(u)
  decomp_kernel<<<dim3(8, 32, 16), 256, 0, stream>>>(u, d1w, d1b, r1);

  // 7) h1 = gelu(r1 @ conv1_w^T)   [32768,512]x[512,2048] -> f16
  gemm_wmma<float, _Float16, false, 1><<<dim3(16, 256, 1), 256, 0, stream>>>(
      r1, c1, nullptr, nullptr, h1, 512, 0, 512, 0, 512, 0, 2048);

  // 8) z = r1 + h1 @ conv2_w^T     (writes into u, dead after step 6)
  gemm_wmma<_Float16, float, false, 0><<<dim3(4, 256, 1), 256, 0, stream>>>(
      h1, c2, nullptr, r1, u, 2048, 0, 2048, 0, 2048, 0, 512);

  // 9) out = decomp(z)
  decomp_kernel<<<dim3(8, 32, 16), 256, 0, stream>>>(u, d2w, d2b,
                                                     (float*)d_out);
}
